// MSR_50130858279882
// MI455X (gfx1250) — matrix operand order repaired and checked by reading, not compiled, not run on hardware
//
#include <hip/hip_runtime.h>
#include <math.h>

// ---------------------------------------------------------------------------
// Multi-scale retinex: out = log1p(x) - (1/3) * sum_s log1p(G_s X G_s)
// G_s = 512x512 symmetric banded Gaussian matrix (zero padding implicit).
// Pass 1: U = (X * G)^T   (NT GEMM, epilogue writes transposed, vectorized)
// Pass 2: S (+)= log1p(G * U^T)  (NT GEMM, B fragments = rows of U)
// Both passes consume B as K-contiguous rows => zero transpose scatters.
// WMMA: v_wmma_f32_16x16x32_bf16 with hi/lo bf16 split (3 WMMAs / product)
// for ~fp32 accuracy at matrix-pipe throughput.
// Staging is software-pipelined: global loads for tile kt+1 are issued before
// the compute barrier of tile kt, so WMMA work hides the HBM/L2 latency.
// ---------------------------------------------------------------------------

typedef __bf16 bf16_t;
typedef __attribute__((ext_vector_type(16))) __bf16 v16bf;
typedef __attribute__((ext_vector_type(8)))  __bf16 v8bf;
typedef __attribute__((ext_vector_type(4)))  __bf16 v4bf;
typedef __attribute__((ext_vector_type(8)))  float  v8f;

#define IMG     512
#define NIMG    24                    // 8 batch * 3 channels
#define TILE_M  128
#define TILE_N  64
#define TILE_K  32
#define LDA     40                    // padded LDS row stride (elements)
#define KSTEPS  (IMG / TILE_K)        // 16
#define IMG_ELEMS ((long)IMG * IMG)   // 262144

__device__ __forceinline__ void split_bf16(float v, bf16_t& h, bf16_t& l) {
  h = (bf16_t)v;                 // RNE to bf16
  l = (bf16_t)(v - (float)h);    // residual
}

__device__ __forceinline__ v16bf pack16(v8bf a, v8bf b) {
  v16bf r;
#pragma unroll
  for (int i = 0; i < 8; ++i) { r[i] = a[i]; r[8 + i] = b[i]; }
  return r;
}

// ---------------------------------------------------------------------------
// Elementwise kernels (vectorized; nElem divisible by 1024)
// ---------------------------------------------------------------------------
__global__ void split_x_kernel(const float4* __restrict__ x,
                               bf16_t* __restrict__ xhi,
                               bf16_t* __restrict__ xlo, int n4) {
  int i = blockIdx.x * blockDim.x + threadIdx.x;
  if (i < n4) {
    float4 v = x[i];
    bf16_t h0, l0, h1, l1, h2, l2, h3, l3;
    split_bf16(v.x, h0, l0);
    split_bf16(v.y, h1, l1);
    split_bf16(v.z, h2, l2);
    split_bf16(v.w, h3, l3);
    v4bf h, l;
    h[0] = h0; h[1] = h1; h[2] = h2; h[3] = h3;
    l[0] = l0; l[1] = l1; l[2] = l2; l[3] = l3;
    *(v4bf*)(xhi + (long)i * 4) = h;
    *(v4bf*)(xlo + (long)i * 4) = l;
  }
}

// G[i][j] = exp(-(j-i)^2/(2 s^2)) / norm for |j-i| <= 2s, else 0
__global__ void build_g_kernel(bf16_t* __restrict__ ghi, bf16_t* __restrict__ glo,
                               float inv2s2, int halfw, float inv_norm) {
  int i = blockIdx.x * blockDim.x + threadIdx.x;   // 512*512 threads
  int r = i >> 9;
  int c = i & (IMG - 1);
  int d = c - r;
  float v = 0.0f;
  if (d >= -halfw && d <= halfw)
    v = expf(-(float)(d * d) * inv2s2) * inv_norm;
  bf16_t h, l;
  split_bf16(v, h, l);
  ghi[i] = h; glo[i] = l;
}

__global__ void final_kernel(const float4* __restrict__ x,
                             const float4* __restrict__ S,
                             float4* __restrict__ out, int n4) {
  int i = blockIdx.x * blockDim.x + threadIdx.x;
  if (i < n4) {
    float4 xv = x[i];
    float4 sv = S[i];
    float4 o;
    o.x = log1pf(xv.x) - sv.x * (1.0f / 3.0f);
    o.y = log1pf(xv.y) - sv.y * (1.0f / 3.0f);
    o.z = log1pf(xv.z) - sv.z * (1.0f / 3.0f);
    o.w = log1pf(xv.w) - sv.w * (1.0f / 3.0f);
    out[i] = o;
  }
}

// ---------------------------------------------------------------------------
// NT WMMA GEMM: C(512x512) = A(512x512) * BT(512x512)^T, batched on blockIdx.z.
// A row-major (M x K); BT row-major (N x K) so B fragments are contiguous.
// A, BT given as bf16 hi/lo pairs; f32 accumulate; 3 WMMAs per K-step product.
// MODE 0: write C TRANSPOSED, split into (Uhi, Ulo) bf16 (vector b128 stores).
// MODE 1: S  = log1p(C)   (natural layout)
// MODE 2: S += log1p(C)
// Workgroup: 256 threads = 8 waves (4x2), tile 128x64, each wave 32x32.
// ---------------------------------------------------------------------------
template <int MODE>
__global__ __launch_bounds__(256)
void gemm_nt_kernel(const bf16_t* __restrict__ Ahi, const bf16_t* __restrict__ Alo,
                    long aStride,
                    const bf16_t* __restrict__ BThi, const bf16_t* __restrict__ BTlo,
                    long bStride,
                    bf16_t* __restrict__ Uhi, bf16_t* __restrict__ Ulo,
                    float* __restrict__ S) {
  __shared__ __align__(16) bf16_t As_hi[TILE_M * LDA];
  __shared__ __align__(16) bf16_t As_lo[TILE_M * LDA];
  __shared__ __align__(16) bf16_t Bs_hi[TILE_N * LDA];
  __shared__ __align__(16) bf16_t Bs_lo[TILE_N * LDA];

  const int tid  = threadIdx.x;
  const int lane = tid & 31;
  const int wave = tid >> 5;
  const int img  = blockIdx.z;
  const int blockRow = blockIdx.y * TILE_M;
  const int blockCol = blockIdx.x * TILE_N;

  const bf16_t* aHiB = Ahi  + (long)img * aStride;
  const bf16_t* aLoB = Alo  + (long)img * aStride;
  const bf16_t* bHiB = BThi + (long)img * bStride;
  const bf16_t* bLoB = BTlo + (long)img * bStride;

  const int waveRow = (wave >> 1) * 32;   // 0,32,64,96
  const int waveCol = (wave & 1) * 32;    // 0,32
  const int m    = lane & 15;
  const int half = lane >> 4;

  v8f acc[2][2];
#pragma unroll
  for (int mi = 0; mi < 2; ++mi)
#pragma unroll
    for (int ni = 0; ni < 2; ++ni)
#pragma unroll
      for (int e = 0; e < 8; ++e) acc[mi][ni][e] = 0.0f;

  // A staging: 128 rows x 4 chunks(8 bf16) = 512 chunks, 2 per thread
  const int ar0 = tid >> 2;
  const int ak0 = (tid & 3) * 8;
  const int ar1 = (tid + 256) >> 2;
  const int ak1 = ((tid + 256) & 3) * 8;
  // B staging: 64 rows x 4 chunks = 256 chunks, 1 per thread
  const int bn  = tid >> 2;
  const int bk  = (tid & 3) * 8;

  // Per-thread base pointers for the staged chunks (K offset added per step)
  const bf16_t* pA0h = aHiB + (long)(blockRow + ar0) * IMG + ak0;
  const bf16_t* pA0l = aLoB + (long)(blockRow + ar0) * IMG + ak0;
  const bf16_t* pA1h = aHiB + (long)(blockRow + ar1) * IMG + ak1;
  const bf16_t* pA1l = aLoB + (long)(blockRow + ar1) * IMG + ak1;
  const bf16_t* pBh  = bHiB + (long)(blockCol + bn) * IMG + bk;
  const bf16_t* pBl  = bLoB + (long)(blockCol + bn) * IMG + bk;

  // Software pipeline: tile registers hold global data for step kt while LDS
  // holds step kt-1. Loads for kt+1 issue before the compute barrier of kt.
  v8bf rA0h = *(const v8bf*)pA0h;
  v8bf rA0l = *(const v8bf*)pA0l;
  v8bf rA1h = *(const v8bf*)pA1h;
  v8bf rA1l = *(const v8bf*)pA1l;
  v8bf rBh  = *(const v8bf*)pBh;
  v8bf rBl  = *(const v8bf*)pBl;

  for (int kt = 0; kt < KSTEPS; ++kt) {
    __syncthreads();                    // previous compute done; LDS reusable
    *(v8bf*)&As_hi[ar0 * LDA + ak0] = rA0h;
    *(v8bf*)&As_lo[ar0 * LDA + ak0] = rA0l;
    *(v8bf*)&As_hi[ar1 * LDA + ak1] = rA1h;
    *(v8bf*)&As_lo[ar1 * LDA + ak1] = rA1l;
    *(v8bf*)&Bs_hi[bn  * LDA + bk ] = rBh;
    *(v8bf*)&Bs_lo[bn  * LDA + bk ] = rBl;

    if (kt + 1 < KSTEPS) {              // issue next tile's loads now; the
      const long kOff = (long)(kt + 1) * TILE_K;   // WMMA phase hides latency
      rA0h = *(const v8bf*)(pA0h + kOff);
      rA0l = *(const v8bf*)(pA0l + kOff);
      rA1h = *(const v8bf*)(pA1h + kOff);
      rA1l = *(const v8bf*)(pA1l + kOff);
      rBh  = *(const v8bf*)(pBh  + kOff);
      rBl  = *(const v8bf*)(pBl  + kOff);
      if (kt + 2 < KSTEPS) {            // hint kt+2 toward cache
        __builtin_prefetch(pA0h + kOff + TILE_K, 0, 3);
        __builtin_prefetch(pA1h + kOff + TILE_K, 0, 3);
        __builtin_prefetch(pBh  + kOff + TILE_K, 0, 3);
        __builtin_prefetch(pBl  + kOff + TILE_K, 0, 3);
      }
    }
    __syncthreads();                    // LDS tile kt visible to all waves

    // Fragment loads per ISA layout tables (wave32).
    v16bf aH[2], aL[2], bH[2], bL[2];
#pragma unroll
    for (int mi = 0; mi < 2; ++mi) {
      const int rowb = (waveRow + mi * 16 + m) * LDA;
      // elems 0..7 = K(half*8 .. +7), elems 8..15 = K(16+half*8 .. +7)
      aH[mi] = pack16(*(const v8bf*)&As_hi[rowb + half * 8],
                      *(const v8bf*)&As_hi[rowb + 16 + half * 8]);
      aL[mi] = pack16(*(const v8bf*)&As_lo[rowb + half * 8],
                      *(const v8bf*)&As_lo[rowb + 16 + half * 8]);
    }
#pragma unroll
    for (int ni = 0; ni < 2; ++ni) {
      const int nb = (waveCol + ni * 16 + m) * LDA + half * 8;
      // lane column n, elems 0..7 = K(half*8 .. +7), elems 8..15 = K(16+half*8 .. +7)
      bH[ni] = pack16(*(const v8bf*)&Bs_hi[nb],
                      *(const v8bf*)&Bs_hi[nb + 16]);
      bL[ni] = pack16(*(const v8bf*)&Bs_lo[nb],
                      *(const v8bf*)&Bs_lo[nb + 16]);
    }

#pragma unroll
    for (int mi = 0; mi < 2; ++mi)
#pragma unroll
      for (int ni = 0; ni < 2; ++ni) {
        acc[mi][ni] = __builtin_amdgcn_wmma_f32_16x16x32_bf16(
            false, aH[mi], false, bH[ni], (short)0, acc[mi][ni], false, false);
        acc[mi][ni] = __builtin_amdgcn_wmma_f32_16x16x32_bf16(
            false, aH[mi], false, bL[ni], (short)0, acc[mi][ni], false, false);
        acc[mi][ni] = __builtin_amdgcn_wmma_f32_16x16x32_bf16(
            false, aL[mi], false, bH[ni], (short)0, acc[mi][ni], false, false);
      }
  }

  // Epilogue. D layout: elem r -> row = half*8 + r, col = lane&15 of 16x16 tile.
  const long base = (long)img * IMG_ELEMS;
#pragma unroll
  for (int mi = 0; mi < 2; ++mi) {
#pragma unroll
    for (int ni = 0; ni < 2; ++ni) {
      const int colg = blockCol + waveCol + ni * 16 + m;
      const int row0 = blockRow + waveRow + mi * 16 + half * 8;
      if (MODE == 0) {
        // transposed vector store: U[colg][row0 .. row0+7]
        v8bf h, l;
#pragma unroll
        for (int r = 0; r < 8; ++r) {
          bf16_t hh, ll;
          split_bf16(acc[mi][ni][r], hh, ll);
          h[r] = hh; l[r] = ll;
        }
        const long uidx = base + (long)colg * IMG + row0;
        *(v8bf*)&Uhi[uidx] = h;
        *(v8bf*)&Ulo[uidx] = l;
      } else {
#pragma unroll
        for (int r = 0; r < 8; ++r) {
          const long idx = base + (long)(row0 + r) * IMG + colg;
          const float y = log1pf(acc[mi][ni][r]);
          S[idx] = (MODE == 1) ? y : (S[idx] + y);
        }
      }
    }
  }
}

// ---------------------------------------------------------------------------
// Host launcher
// ---------------------------------------------------------------------------
extern "C" void kernel_launch(void* const* d_in, const int* in_sizes, int n_in,
                              void* d_out, int out_size, void* d_ws, size_t ws_size,
                              hipStream_t stream) {
  const float* x = (const float*)d_in[0];
  float* out = (float*)d_out;

  const long nElem = (long)NIMG * IMG_ELEMS;   // 6,291,456

  // Workspace layout
  char* ws = (char*)d_ws;
  size_t off = 0;
  bf16_t* Xhi = (bf16_t*)(ws + off); off += (size_t)nElem * 2;
  bf16_t* Xlo = (bf16_t*)(ws + off); off += (size_t)nElem * 2;
  bf16_t* Ghi = (bf16_t*)(ws + off); off += (size_t)IMG_ELEMS * 2;
  bf16_t* Glo = (bf16_t*)(ws + off); off += (size_t)IMG_ELEMS * 2;
  bf16_t* Uhi = (bf16_t*)(ws + off); off += (size_t)nElem * 2;
  bf16_t* Ulo = (bf16_t*)(ws + off); off += (size_t)nElem * 2;
  float*  Sac = (float*) (ws + off); off += (size_t)nElem * 4;
  if (ws_size < off) return;   // workspace too small; nothing sane to do

  const int n4 = (int)(nElem / 4);
  split_x_kernel<<<dim3((n4 + 255) / 256), dim3(256), 0, stream>>>(
      (const float4*)x, Xhi, Xlo, n4);

  const int scales[3] = {15, 80, 250};
  const dim3 gemmGrid(IMG / TILE_N, IMG / TILE_M, NIMG);   // (8,4,24)
  const dim3 gemmBlock(256);

  for (int si = 0; si < 3; ++si) {
    const int s = scales[si];
    const int halfw = 2 * s;
    double norm = 0.0;
    for (int t = -halfw; t <= halfw; ++t)
      norm += exp(-(double)(t * t) / (2.0 * (double)s * (double)s));
    const float inv_norm = (float)(1.0 / norm);
    const float inv2s2   = (float)(1.0 / (2.0 * (double)s * (double)s));

    build_g_kernel<<<dim3((IMG * IMG) / 256), dim3(256), 0, stream>>>(
        Ghi, Glo, inv2s2, halfw, inv_norm);

    // Pass 1 (horizontal): U = (X * G)^T. A = X (batched), BT = G (symmetric).
    gemm_nt_kernel<0><<<gemmGrid, gemmBlock, 0, stream>>>(
        Xhi, Xlo, IMG_ELEMS, Ghi, Glo, 0L, Uhi, Ulo, nullptr);

    // Pass 2 (vertical): S (+)= log1p(G * U^T). A = G, BT = U (batched).
    if (si == 0)
      gemm_nt_kernel<1><<<gemmGrid, gemmBlock, 0, stream>>>(
          Ghi, Glo, 0L, Uhi, Ulo, IMG_ELEMS, nullptr, nullptr, Sac);
    else
      gemm_nt_kernel<2><<<gemmGrid, gemmBlock, 0, stream>>>(
          Ghi, Glo, 0L, Uhi, Ulo, IMG_ELEMS, nullptr, nullptr, Sac);
  }

  final_kernel<<<dim3((n4 + 255) / 256), dim3(256), 0, stream>>>(
      (const float4*)x, (const float4*)Sac, (float4*)out, n4);
}
